// MultiHeadSelfAttention_87754771792167
// MI455X (gfx1250) — matrix operand order repaired and checked by reading, not compiled, not run on hardware
//
#include <hip/hip_runtime.h>
#include <math.h>
#include <stdint.h>

// ---------------------------------------------------------------------------
// MI455X (gfx1250) CDNA5 flash-attention MHA.
// B=1, S=4096, E=1024, H=16, HD=64.  All GEMMs via v_wmma_f32_16x16x32_bf16.
// K/V tiles staged to LDS with async-to-LDS DMA, double buffered.
// ---------------------------------------------------------------------------

#define SEQ 4096
#define EMB 1024
#define NH  16
#define HD  64

typedef __attribute__((ext_vector_type(16))) __bf16 bf16x16;
typedef __attribute__((ext_vector_type(8)))  __bf16 bf16x8;
typedef __attribute__((ext_vector_type(8)))  float  f32x8;
typedef int i32x4_t __attribute__((vector_size(16)));

__device__ inline f32x8 wmma_bf16(bf16x16 a, bf16x16 b, f32x8 c) {
  return __builtin_amdgcn_wmma_f32_16x16x32_bf16(
      /*neg_a=*/false, a, /*neg_b=*/false, b,
      /*c_mod=*/(short)0, c, /*reuse_a=*/false, /*reuse_b=*/false);
}

__device__ inline f32x8 zero8() {
  f32x8 z;
#pragma unroll
  for (int i = 0; i < 8; i++) z[i] = 0.0f;
  return z;
}

// -------- async global->LDS copy (16B), ASYNCcnt-tracked -------------------
__device__ inline void cp_async16(void* lds_dst, const void* gsrc) {
#if __has_builtin(__builtin_amdgcn_global_load_async_to_lds_b128)
  __builtin_amdgcn_global_load_async_to_lds_b128(
      (__attribute__((address_space(1))) i32x4_t*)(uintptr_t)gsrc,
      (__attribute__((address_space(3))) i32x4_t*)(unsigned)(uintptr_t)lds_dst,
      0, 0);
#else
  *(int4*)lds_dst = *(const int4*)gsrc;   // sync fallback
#endif
}

__device__ inline void wait_async() {
#if __has_builtin(__builtin_amdgcn_s_wait_asynccnt)
  __builtin_amdgcn_s_wait_asynccnt(0);
#endif
}

// -------- fragment loaders (ISA 7.12.2 layouts, wave32) ---------------------
// A (16x32, 16-bit): lane m=lane&15, hk=lane>>4.
//   a[e] = row[m][ kbase + (e<8 ? hk*8+e : 16+hk*8+(e-8)) ]  -> two 16B chunks
__device__ inline bf16x16 ld_a16(const __bf16* tile, int stride, int kbase, int lane) {
  int m = lane & 15, hk = (lane >> 4) & 1;
  const __bf16* p0 = tile + (size_t)m * stride + kbase + hk * 8;
  bf16x8 c0 = *(const bf16x8*)(p0);
  bf16x8 c1 = *(const bf16x8*)(p0 + 16);
  bf16x16 a;
#pragma unroll
  for (int i = 0; i < 8; i++) { a[i] = c0[i]; a[i + 8] = c1[i]; }
  return a;
}

__device__ inline bf16x16 ld_a32(const float* tile, int stride, int kbase, int lane) {
  int m = lane & 15, hk = (lane >> 4) & 1;
  const float* p0 = tile + (size_t)m * stride + kbase + hk * 8;
  bf16x16 a;
#pragma unroll
  for (int i = 0; i < 8; i++) { a[i] = (__bf16)p0[i]; a[i + 8] = (__bf16)p0[16 + i]; }
  return a;
}

// B (32x16, 16-bit): lane n=lane&15, hk=lane>>4.  b[e] = B[kbase + (e<8 ? hk*8+e : 16+hk*8+(e-8))][n].
// Source layout is "n-major rows, contiguous in k" -> two 16B chunks.
__device__ inline bf16x16 ld_b16(const __bf16* base, int stride, int kbase, int lane) {
  int n = lane & 15, hk = (lane >> 4) & 1;
  const __bf16* p = base + (size_t)n * stride + kbase + hk * 8;
  bf16x8 c0 = *(const bf16x8*)(p);
  bf16x8 c1 = *(const bf16x8*)(p + 16);
  bf16x16 b;
#pragma unroll
  for (int i = 0; i < 8; i++) { b[i] = c0[i]; b[i + 8] = c1[i]; }
  return b;
}

__device__ inline bf16x16 ld_b32(const float* base, int stride, int kbase, int lane) {
  int n = lane & 15, hk = (lane >> 4) & 1;
  const float* p = base + (size_t)n * stride + kbase + hk * 8;
  bf16x16 b;
#pragma unroll
  for (int i = 0; i < 8; i++) { b[i] = (__bf16)p[i]; b[i + 8] = (__bf16)p[16 + i]; }
  return b;
}

// ---------------------------------------------------------------------------
// Kernel 1: per-head QKV projections.  Q pre-scaled by 1/sqrt(HD).
// Q,K stored [H][S][HD] bf16 row-major; V stored transposed [H][HD][S] bf16.
// grid = (SEQ/128, NH), block = 256 (8 waves, one 16-row tile each)
// ---------------------------------------------------------------------------
__global__ __launch_bounds__(256) void qkv_kernel(
    const float* __restrict__ x,  const float* __restrict__ wq,
    const float* __restrict__ wk, const float* __restrict__ wv,
    __bf16* __restrict__ Qb, __bf16* __restrict__ Kb, __bf16* __restrict__ Vt) {
  int lane = threadIdx.x & 31;
  int wave = threadIdx.x >> 5;
  int h  = blockIdx.y;
  int st = (blockIdx.x * 8 + wave) * 16;          // row tile base
  int n  = lane & 15, hk = (lane >> 4) & 1;

  const float* xt = x + (size_t)st * EMB + h * HD;   // head slice, stride EMB
  bf16x16 xa0 = ld_a32(xt, EMB, 0,  lane);
  bf16x16 xa1 = ld_a32(xt, EMB, 32, lane);

  const float* Wq = wq + (size_t)h * HD * HD;
  const float* Wk = wk + (size_t)h * HD * HD;
  const float* Wv = wv + (size_t)h * HD * HD;

#pragma unroll
  for (int nb = 0; nb < HD; nb += 16) {
    // Q (scaled by 1/sqrt(64))
    f32x8 c = zero8();
    c = wmma_bf16(xa0, ld_b32(Wq + nb * HD, HD, 0,  lane), c);
    c = wmma_bf16(xa1, ld_b32(Wq + nb * HD, HD, 32, lane), c);
#pragma unroll
    for (int r = 0; r < 8; r++)
      Qb[(size_t)h * SEQ * HD + (size_t)(st + r + 8 * hk) * HD + nb + n] =
          (__bf16)(c[r] * 0.125f);
    // K
    c = zero8();
    c = wmma_bf16(xa0, ld_b32(Wk + nb * HD, HD, 0,  lane), c);
    c = wmma_bf16(xa1, ld_b32(Wk + nb * HD, HD, 32, lane), c);
#pragma unroll
    for (int r = 0; r < 8; r++)
      Kb[(size_t)h * SEQ * HD + (size_t)(st + r + 8 * hk) * HD + nb + n] = (__bf16)c[r];
    // V  (stored transposed: Vt[h][feature][s])
    c = zero8();
    c = wmma_bf16(xa0, ld_b32(Wv + nb * HD, HD, 0,  lane), c);
    c = wmma_bf16(xa1, ld_b32(Wv + nb * HD, HD, 32, lane), c);
#pragma unroll
    for (int r = 0; r < 8; r++)
      Vt[(size_t)h * HD * SEQ + (size_t)(nb + n) * SEQ + st + r + 8 * hk] = (__bf16)c[r];
  }
}

// ---------------------------------------------------------------------------
// Kernel 2: causal flash attention.  One wave owns a 16x64 Q strip; the 4
// waves of a block share async-staged K/V tiles (double buffered in LDS).
// grid = (SEQ/64, NH), block = 128 (4 waves).
// ---------------------------------------------------------------------------
__global__ __launch_bounds__(128) void attn_kernel(
    const __bf16* __restrict__ Qb, const __bf16* __restrict__ Kb,
    const __bf16* __restrict__ Vt, __bf16* __restrict__ Ob) {
  __shared__ __align__(16) __bf16 kstage[2][32][HD];   // 2 x 4KB : K rows
  __shared__ __align__(16) __bf16 vstage[2][HD][32];   // 2 x 4KB : V^T rows
  __shared__ __align__(16) __bf16 pstage[4][16][32];   // C->A relayout staging

  int tid  = threadIdx.x;
  int lane = tid & 31;
  int wave = tid >> 5;
  int h  = blockIdx.y;
  int q0 = (blockIdx.x * 4 + wave) * 16;
  int n  = lane & 15, hk = (lane >> 4) & 1;

  const __bf16* Qh = Qb + (size_t)h * SEQ * HD;
  const __bf16* Kh = Kb + (size_t)h * SEQ * HD;
  const __bf16* Vh = Vt + (size_t)h * HD * SEQ;
  __bf16* ps = &pstage[wave][0][0];

  bf16x16 qa0 = ld_a16(Qh + (size_t)q0 * HD, HD, 0,  lane);
  bf16x16 qa1 = ld_a16(Qh + (size_t)q0 * HD, HD, 32, lane);

  f32x8 o0 = zero8(), o1 = zero8(), o2 = zero8(), o3 = zero8();
  float mrow[8], lrow[8];
#pragma unroll
  for (int r = 0; r < 8; r++) { mrow[r] = -INFINITY; lrow[r] = 0.0f; }

  int ktend     = q0 + 16;                 // this wave's causal bound
  int ktend_blk = blockIdx.x * 64 + 64;    // block-uniform bound
  int ntiles    = ktend_blk / 32;

  // stage a 32-key tile (K: 4KB contiguous; V^T: 64 rows x 64B) into buf
  auto stage = [&](int buf, int kt) {
    const char* gk = (const char*)(Kh + (size_t)kt * HD);
    char* lk = (char*)&kstage[buf][0][0];
    char* lv = (char*)&vstage[buf][0][0];
#pragma unroll
    for (int j = 0; j < 2; j++) {
      int c = tid + j * 128;                         // 256 x 16B = 4KB
      cp_async16(lk + c * 16, gk + c * 16);
    }
#pragma unroll
    for (int j = 0; j < 2; j++) {
      int c = tid + j * 128;                         // 256 x 16B = 4KB
      int row = c >> 2, off = (c & 3) * 16;
      cp_async16(lv + c * 16,
                 (const char*)(Vh + (size_t)row * SEQ + kt) + off);
    }
  };

  stage(0, 0);
  wait_async();
  __syncthreads();

  for (int i = 0; i < ntiles; i++) {
    int kt  = i * 32;
    int cur = i & 1;
    if (i + 1 < ntiles) stage(cur ^ 1, kt + 32);     // prefetch next tile

    if (kt < ktend) {                                // wave-uniform predicate
      const __bf16* kb = &kstage[cur][0][0];
      const __bf16* vb = &vstage[cur][0][0];

      // ---- S = Q * K^T for 32 keys (two 16x16 C tiles) ----
      f32x8 s0 = zero8(), s1 = zero8();
      s0 = wmma_bf16(qa0, ld_b16(kb,           HD, 0,  lane), s0);
      s0 = wmma_bf16(qa1, ld_b16(kb,           HD, 32, lane), s0);
      s1 = wmma_bf16(qa0, ld_b16(kb + 16 * HD, HD, 0,  lane), s1);
      s1 = wmma_bf16(qa1, ld_b16(kb + 16 * HD, HD, 32, lane), s1);

      if (kt + 32 > q0) {                            // diagonal: causal mask
#pragma unroll
        for (int r = 0; r < 8; r++) {
          int qr = q0 + r + 8 * hk;
          if (kt + n      > qr) s0[r] = -INFINITY;
          if (kt + 16 + n > qr) s1[r] = -INFINITY;
        }
      }

      // ---- online softmax (row = r + 8*hk; reduce across 16-lane half) ----
#pragma unroll
      for (int r = 0; r < 8; r++) {
        float mx = fmaxf(s0[r], s1[r]);
        mx = fmaxf(mx, __shfl_xor(mx, 1, 32));
        mx = fmaxf(mx, __shfl_xor(mx, 2, 32));
        mx = fmaxf(mx, __shfl_xor(mx, 4, 32));
        mx = fmaxf(mx, __shfl_xor(mx, 8, 32));
        float mn = fmaxf(mrow[r], mx);
        float al = __expf(mrow[r] - mn);
        mrow[r] = mn;
        float p0 = __expf(s0[r] - mn);
        float p1 = __expf(s1[r] - mn);
        float t = p0 + p1;
        t += __shfl_xor(t, 1, 32);
        t += __shfl_xor(t, 2, 32);
        t += __shfl_xor(t, 4, 32);
        t += __shfl_xor(t, 8, 32);
        lrow[r] = al * lrow[r] + t;
        o0[r] *= al; o1[r] *= al; o2[r] *= al; o3[r] *= al;
        ps[(r + 8 * hk) * 32 + n]      = (__bf16)p0;
        ps[(r + 8 * hk) * 32 + 16 + n] = (__bf16)p1;
      }
      __threadfence_block();                         // DS write->read ordering

      // ---- O += P(16x32) * V(32x64) ----
      bf16x16 pa = ld_a16(ps, 32, 0, lane);
      o0 = wmma_bf16(pa, ld_b16(vb +  0 * 32, 32, 0, lane), o0);
      o1 = wmma_bf16(pa, ld_b16(vb + 16 * 32, 32, 0, lane), o1);
      o2 = wmma_bf16(pa, ld_b16(vb + 32 * 32, 32, 0, lane), o2);
      o3 = wmma_bf16(pa, ld_b16(vb + 48 * 32, 32, 0, lane), o3);
    }

    wait_async();          // staged data for next iteration has landed
    __syncthreads();       // all waves done reading cur, writes visible
  }

  // ---- epilogue: normalize and store attn output [S][E] bf16 ----
#pragma unroll
  for (int r = 0; r < 8; r++) {
    float inv = 1.0f / lrow[r];
    size_t row = (size_t)(q0 + r + 8 * hk) * EMB + h * HD + n;
    Ob[row + 0]  = (__bf16)(o0[r] * inv);
    Ob[row + 16] = (__bf16)(o1[r] * inv);
    Ob[row + 32] = (__bf16)(o2[r] * inv);
    Ob[row + 48] = (__bf16)(o3[r] * inv);
  }
}

// ---------------------------------------------------------------------------
// Kernel 3: out = Ob(S x E, bf16) @ wo^T (wo is [E][E] f32, row=out,col=in)
// grid = (SEQ/16, EMB/64), block = 128: wave w -> output cols blockIdx.y*64+w*16
// ---------------------------------------------------------------------------
__global__ __launch_bounds__(128) void proj_kernel(
    const __bf16* __restrict__ Ob, const float* __restrict__ wo,
    float* __restrict__ out) {
  int lane = threadIdx.x & 31;
  int wave = threadIdx.x >> 5;
  int m0 = blockIdx.x * 16;
  int nb = blockIdx.y * 64 + wave * 16;
  int n  = lane & 15, hk = (lane >> 4) & 1;

  f32x8 c = zero8();
#pragma unroll 4
  for (int kk = 0; kk < EMB; kk += 32) {
    bf16x16 a = ld_a16(Ob + (size_t)m0 * EMB, EMB, kk, lane);
    bf16x16 b = ld_b32(wo + (size_t)nb * EMB, EMB, kk, lane);
    c = wmma_bf16(a, b, c);
  }
#pragma unroll
  for (int r = 0; r < 8; r++)
    out[(size_t)(m0 + r + 8 * hk) * EMB + nb + n] = c[r];
}

// ---------------------------------------------------------------------------
extern "C" void kernel_launch(void* const* d_in, const int* in_sizes, int n_in,
                              void* d_out, int out_size, void* d_ws, size_t ws_size,
                              hipStream_t stream) {
  const float* x  = (const float*)d_in[0];
  const float* wq = (const float*)d_in[1];
  const float* wk = (const float*)d_in[2];
  const float* wv = (const float*)d_in[3];
  const float* wo = (const float*)d_in[4];
  float* out = (float*)d_out;

  char* ws = (char*)d_ws;
  const size_t MB = (size_t)1 << 20;
  __bf16* Qb = (__bf16*)(ws + 0 * MB);   // [NH][SEQ][HD]   8 MiB
  __bf16* Kb = (__bf16*)(ws + 8 * MB);   // [NH][SEQ][HD]   8 MiB
  __bf16* Vt = (__bf16*)(ws + 16 * MB);  // [NH][HD][SEQ]   8 MiB
  __bf16* Ob = (__bf16*)(ws + 24 * MB);  // [SEQ][EMB]      8 MiB

  qkv_kernel<<<dim3(SEQ / 128, NH), 256, 0, stream>>>(x, wq, wk, wv, Qb, Kb, Vt);
  attn_kernel<<<dim3(SEQ / 64, NH), 128, 0, stream>>>(Qb, Kb, Vt, Ob);
  proj_kernel<<<dim3(SEQ / 16, EMB / 64), 128, 0, stream>>>(Ob, wo, out);
}
